// GptOssExpertsNew_29394756173986
// MI455X (gfx1250) — matrix operand order repaired and checked by reading, not compiled, not run on hardware
//
#include <hip/hip_runtime.h>
#include <hip/hip_bf16.h>

// ---------------------------------------------------------------------------
// GPT-OSS MoE experts for MI455X (gfx1250, wave32, WMMA bf16 16x16x32).
// E=8 experts, H=I=2880, T=2048 tokens, top-k=2.
// Routed-only compute: one workgroup per (expert, 16-token tile).
// Weights pre-packed to bf16 WMMA B-fragment layout in workspace (if it fits),
// async global->LDS staging for activations, packed bf16 converts.
// ---------------------------------------------------------------------------

#define HDIM   2880
#define IDIM   2880
#define I2DIM  5760
#define NEXP   8
#define NTOK   2048
#define TILES  128          // NTOK / 16
#define NTILES 180          // HDIM / 16
#define KBLK   90           // HDIM / 32
#define CHROWS 4            // token rows per async staging chunk

// fragment dwords per (e, ntile, k32) block: 32 lanes * 8 dwords
#define FRAG_DW   256
#define STREAM_DW ((size_t)NEXP * NTILES * KBLK * FRAG_DW)   // 33,177,600 dwords

typedef __attribute__((ext_vector_type(16))) __bf16 v16bf;
typedef __attribute__((ext_vector_type(8)))  float  v8f;

union AFrag {
    v16bf    v;
    unsigned u[8];
    uint4    q[2];
};

// fp32x2 -> packed bf16x2 (prefer V_CVT_PK_BF16_F32 when the builtin exists)
__device__ __forceinline__ unsigned cvt_pk_bf16(float lo, float hi) {
#if defined(__gfx1250__) && __has_builtin(__builtin_amdgcn_cvt_pk_bf16_f32)
    typedef __attribute__((ext_vector_type(2))) __bf16 v2bf;
    v2bf r = __builtin_amdgcn_cvt_pk_bf16_f32(lo, hi);
    return __builtin_bit_cast(unsigned, r);
#else
    unsigned a = __float_as_uint(lo);
    unsigned b = __float_as_uint(hi);
    a = (a + 0x7FFFu + ((a >> 16) & 1u)) >> 16;
    b = (b + 0x7FFFu + ((b >> 16) & 1u)) >> 16;
    return a | (b << 16);
#endif
}

__device__ __forceinline__ unsigned lds_addr_of(const void* p) {
    return (unsigned)(uintptr_t)(__attribute__((address_space(3))) const void*)p;
}

// ---------------------------------------------------------------------------
// Small utility kernels
// ---------------------------------------------------------------------------
__global__ void zero_f32_kernel(float* __restrict__ p, int n) {
    int i = blockIdx.x * 256 + threadIdx.x;
    if (i < n) p[i] = 0.0f;
}

__global__ void zero_i32_kernel(int* __restrict__ p, int n) {
    int i = blockIdx.x * 256 + threadIdx.x;
    if (i < n) p[i] = 0;
}

__global__ void router_kernel(const int* __restrict__ ridx,
                              int* __restrict__ counts,
                              int* __restrict__ lists) {
    int t = blockIdx.x * 256 + threadIdx.x;
    if (t >= NTOK) return;
    int e0 = ridx[2 * t + 0];
    int e1 = ridx[2 * t + 1];
    int pos = atomicAdd(&counts[e0], 1);
    lists[e0 * NTOK + pos] = t;
    if (e1 != e0) {                    // reference mask is "any" -> count once
        pos = atomicAdd(&counts[e1], 1);
        lists[e1 * NTOK + pos] = t;
    }
}

// ---------------------------------------------------------------------------
// Weight pre-pack: fp32 [H,cols] -> bf16 WMMA B-fragment stream
//   dst dword id = (((e*NTILES + nt)*KBLK + k32)*32 + lane)*8 + v
//   value = pack_bf16(W[kb][col], W[kb+1][col]),  kb = k32*32 + 2v + 16*(lane>>4)
// mode 0: gate cols (2*(nt*16+n))   of gate_up_w
// mode 1: up   cols (2*(nt*16+n)+1) of gate_up_w
// mode 2: cols (nt*16+n)            of down_w
// ---------------------------------------------------------------------------
__global__ void prepack_kernel(const float* __restrict__ src,
                               unsigned* __restrict__ dst, int mode) {
    size_t id  = (size_t)blockIdx.x * 256 + threadIdx.x;
    int   v    = (int)(id & 7);
    int   lane = (int)((id >> 3) & 31);
    size_t blk = id >> 8;
    int   k32  = (int)(blk % KBLK);
    int   nt   = (int)((blk / KBLK) % NTILES);
    int   e    = (int)(blk / ((size_t)KBLK * NTILES));
    int   n    = lane & 15;
    int   h    = lane >> 4;
    int   kb   = k32 * 32 + 2 * v + (h << 4);
    int   stride = (mode == 2) ? HDIM : I2DIM;
    int   col    = (mode == 2) ? (nt * 16 + n) : (2 * (nt * 16 + n) + mode);
    size_t base  = (size_t)e * HDIM * stride + (size_t)kb * stride + col;
    float lo = src[base];
    float hi = src[base + stride];
    dst[id] = cvt_pk_bf16(lo, hi);
}

// ---------------------------------------------------------------------------
// Fused expert kernel: gate_up GEMM -> clamped SiLU-GLU -> down GEMM -> scatter
// blockDim = 256 (8 waves). One block = (expert e, tile of 16 routed tokens).
// PRE=true : B fragments come from pre-packed bf16 streams (2x b128 per frag)
// PRE=false: B fragments converted from fp32 weights in-loop (fallback)
// ---------------------------------------------------------------------------
template <bool PRE>
__global__ __launch_bounds__(256) void moe_expert_kernel(
    const float* __restrict__ x,     // [T, H]
    const float* __restrict__ guw,   // [E, H, 2I]
    const float* __restrict__ gub,   // [E, 2I]
    const float* __restrict__ dw,    // [E, I, H]
    const float* __restrict__ db,    // [E, H]
    const float* __restrict__ rw,    // [T, E]
    const int*   __restrict__ counts,
    const int*   __restrict__ lists,
    const unsigned* __restrict__ gpk,   // packed gate stream
    const unsigned* __restrict__ upk,   // packed up stream
    const unsigned* __restrict__ dpk,   // packed down stream
    float* __restrict__ out)         // [T, H]
{
    __shared__ float          stage[CHROWS * HDIM];      // fp32 async landing (45 KB)
    __shared__ unsigned       xs32[16 * (HDIM / 2)];     // bf16x2 activations (90 KB)
    __shared__ unsigned short hs[16 * HDIM];             // bf16 intermediate  (90 KB)
    __shared__ int   tok_s[16];
    __shared__ float wt_s[16];

    const int e    = blockIdx.x / TILES;
    const int tile = blockIdx.x % TILES;
    const int cnt  = counts[e];
    if (tile * 16 >= cnt) return;

    const int tid = threadIdx.x;
    if (tid < 16) {
        int r = tile * 16 + tid;
        if (r < cnt) {
            int t = lists[e * NTOK + r];
            tok_s[tid] = t;
            wt_s[tid]  = rw[t * NEXP + e];
        } else {
            tok_s[tid] = -1;
            wt_s[tid]  = 0.0f;
        }
    }
    __syncthreads();

    // ---- stage activation tile: async global->LDS fp32, then pack to bf16 --
    const unsigned stage_base = lds_addr_of(stage);
    for (int c0 = 0; c0 < 16; c0 += CHROWS) {
        for (int u = tid; u < CHROWS * (HDIM / 4); u += 256) {
            int r   = u / (HDIM / 4);
            int col = (u - r * (HDIM / 4)) * 4;
            int t   = tok_s[c0 + r];
            if (t >= 0) {
                const float* g = x + (size_t)t * HDIM + col;
                unsigned laddr = stage_base + (unsigned)((r * HDIM + col) * 4);
                asm volatile("global_load_async_to_lds_b128 %0, %1, off"
                             :: "v"(laddr), "v"(g) : "memory");
            } else {
                float4 z = {0.0f, 0.0f, 0.0f, 0.0f};
                *(float4*)&stage[r * HDIM + col] = z;
            }
        }
        asm volatile("s_wait_asynccnt 0x0" ::: "memory");
        __syncthreads();
        for (int u = tid; u < CHROWS * (HDIM / 2); u += 256) {
            int r  = u / (HDIM / 2);
            int cp = u - r * (HDIM / 2);
            xs32[(c0 + r) * (HDIM / 2) + cp] =
                cvt_pk_bf16(stage[r * HDIM + 2 * cp], stage[r * HDIM + 2 * cp + 1]);
        }
        __syncthreads();   // stage reused next chunk
    }

    const int lane  = tid & 31;
    const int wv    = tid >> 5;
    const int halfb = (lane >> 4) & 1;
    const int nidx  = lane & 15;
    const int abase = nidx * (HDIM / 2);

    // =======================================================================
    // Phase 1: gate_up projection + activation -> hs
    // =======================================================================
    const float* wg = guw + (size_t)e * HDIM * I2DIM;
    for (int nt = wv; nt < NTILES; nt += 8) {
        const int f0   = nt * 16;
        const int colg = 2 * (f0 + nidx);
        v8f accg = {};
        v8f accu = {};
        const unsigned* gfrag =
            gpk + ((size_t)((e * NTILES + nt) * KBLK) * 32 + lane) * 8;
        const unsigned* ufrag =
            upk + ((size_t)((e * NTILES + nt) * KBLK) * 32 + lane) * 8;
        for (int kk = 0; kk < HDIM; kk += 32) {
            AFrag a, bg, bu;
            // A fragment: 2x ds_load_b128 (contiguous k = 16*half + s, matches B)
            a.q[0] = *(const uint4*)&xs32[abase + (kk >> 1) + 8 * halfb];
            a.q[1] = *(const uint4*)&xs32[abase + (kk >> 1) + 4 + 8 * halfb];
            if constexpr (PRE) {
                bg.q[0] = *(const uint4*)(gfrag + 0);
                bg.q[1] = *(const uint4*)(gfrag + 4);
                bu.q[0] = *(const uint4*)(ufrag + 0);
                bu.q[1] = *(const uint4*)(ufrag + 4);
                gfrag += FRAG_DW;
                ufrag += FRAG_DW;
            } else {
#pragma unroll
                for (int v = 0; v < 8; ++v) {
                    int kb = kk + 2 * v + (halfb << 4);
                    const float2 lo = *(const float2*)(wg + (size_t)kb * I2DIM + colg);
                    const float2 hi = *(const float2*)(wg + (size_t)(kb + 1) * I2DIM + colg);
                    bg.u[v] = cvt_pk_bf16(lo.x, hi.x);
                    bu.u[v] = cvt_pk_bf16(lo.y, hi.y);
                }
            }
            accg = __builtin_amdgcn_wmma_f32_16x16x32_bf16(
                false, a.v, false, bg.v, (short)0, accg, false, false);
            accu = __builtin_amdgcn_wmma_f32_16x16x32_bf16(
                false, a.v, false, bu.v, (short)0, accu, false, false);
        }
        const float bgs = gub[e * I2DIM + colg];
        const float bus = gub[e * I2DIM + colg + 1];
#pragma unroll
        for (int j = 0; j < 8; ++j) {
            float g = accg[j] + bgs;
            float u = accu[j] + bus;
            g = fminf(g, 7.0f);
            u = fminf(fmaxf(u, -7.0f), 7.0f);
            float glu = g * __builtin_amdgcn_rcpf(1.0f + __expf(-1.702f * g));
            float hv  = (u + 1.0f) * glu;
            int m = j + (halfb << 3);
            hs[m * HDIM + f0 + nidx] = (unsigned short)cvt_pk_bf16(hv, hv);
        }
    }
    __syncthreads();

    // =======================================================================
    // Phase 2: down projection + routing-weight scatter
    // =======================================================================
    const float* dwe = dw + (size_t)e * IDIM * HDIM;
    for (int nt = wv; nt < NTILES; nt += 8) {
        const int col = nt * 16 + nidx;
        v8f acc = {};
        const unsigned* dfrag =
            dpk + ((size_t)((e * NTILES + nt) * KBLK) * 32 + lane) * 8;
#pragma unroll 2
        for (int kk = 0; kk < IDIM; kk += 32) {
            AFrag a, b;
            a.q[0] = *(const uint4*)&hs[nidx * HDIM + kk + (halfb << 4)];
            a.q[1] = *(const uint4*)&hs[nidx * HDIM + kk + 8 + (halfb << 4)];
            if constexpr (PRE) {
                b.q[0] = *(const uint4*)(dfrag + 0);
                b.q[1] = *(const uint4*)(dfrag + 4);
                dfrag += FRAG_DW;
            } else {
#pragma unroll
                for (int v = 0; v < 8; ++v) {
                    int kb = kk + 2 * v + (halfb << 4);
                    const float* wp = dwe + (size_t)kb * HDIM + col;
                    b.u[v] = cvt_pk_bf16(wp[0], wp[HDIM]);
                }
            }
            acc = __builtin_amdgcn_wmma_f32_16x16x32_bf16(
                false, a.v, false, b.v, (short)0, acc, false, false);
        }
        const float bias = db[e * HDIM + col];
#pragma unroll
        for (int j = 0; j < 8; ++j) {
            int m = j + (halfb << 3);
            int t = tok_s[m];
            if (t >= 0) {
                float val = wt_s[m] * (acc[j] + bias);
                atomicAdd(&out[(size_t)t * HDIM + col], val);
            }
        }
    }
}

// ---------------------------------------------------------------------------
// Launch
// ---------------------------------------------------------------------------
extern "C" void kernel_launch(void* const* d_in, const int* in_sizes, int n_in,
                              void* d_out, int out_size, void* d_ws, size_t ws_size,
                              hipStream_t stream) {
    const float* x    = (const float*)d_in[0];   // hidden_states [2,1024,2880]
    const float* guw  = (const float*)d_in[1];   // gate_up_w     [8,2880,5760]
    const float* gub  = (const float*)d_in[2];   // gate_up_b     [8,5760]
    const float* dw   = (const float*)d_in[3];   // down_w        [8,2880,2880]
    const float* db   = (const float*)d_in[4];   // down_b        [8,2880]
    const int*   ridx = (const int*)d_in[5];     // router_indices[2048,2]
    const float* rw   = (const float*)d_in[6];   // routing_weights[2048,8]
    float* out = (float*)d_out;                  // [2,1024,2880] fp32

    int* counts = (int*)d_ws;                              // [8]
    int* lists  = (int*)((char*)d_ws + 4096);              // [8,2048]
    unsigned* gpk = (unsigned*)((char*)d_ws + (128 << 10));
    unsigned* upk = gpk + STREAM_DW;
    unsigned* dpk = upk + STREAM_DW;
    const size_t needed = (size_t)(128 << 10) + 3 * STREAM_DW * 4;
    const bool pre = (ws_size >= needed);

    const int out_elems = NTOK * HDIM;
    zero_f32_kernel<<<(out_elems + 255) / 256, 256, 0, stream>>>(out, out_elems);
    zero_i32_kernel<<<1, 256, 0, stream>>>(counts, NEXP);
    router_kernel<<<(NTOK + 255) / 256, 256, 0, stream>>>(ridx, counts, lists);

    if (pre) {
        const int pp_blocks = (int)(STREAM_DW / 256);      // 129,600
        prepack_kernel<<<pp_blocks, 256, 0, stream>>>(guw, gpk, 0);
        prepack_kernel<<<pp_blocks, 256, 0, stream>>>(guw, upk, 1);
        prepack_kernel<<<pp_blocks, 256, 0, stream>>>(dw,  dpk, 2);
        moe_expert_kernel<true><<<NEXP * TILES, 256, 0, stream>>>(
            x, guw, gub, dw, db, rw, counts, lists, gpk, upk, dpk, out);
    } else {
        moe_expert_kernel<false><<<NEXP * TILES, 256, 0, stream>>>(
            x, guw, gub, dw, db, rw, counts, lists, gpk, upk, dpk, out);
    }
}
